// FlashAttention_28089086116066
// MI455X (gfx1250) — matrix operand order repaired and checked by reading, not compiled, not run on hardware
//
#include <hip/hip_runtime.h>
#include <hip/hip_fp16.h>

typedef __attribute__((ext_vector_type(16))) _Float16 v16h;
typedef __attribute__((ext_vector_type(8)))  float    v8f;
typedef __attribute__((ext_vector_type(4)))  int      v4i;

#define B_    2
#define S_    2048
#define HID_  4096
#define HQ_   32
#define HKV_  8
#define D_    128

union AFrag { v16h v; _Float16 e[16]; };
union CFrag { v8f  v; float    e[8];  };
union Pack8 { uint4 u; _Float16 h[8]; };

// CDNA5 async global->LDS path (ASYNCcnt). Guarded so either toolchain compiles.
// Builtin signature (from probe): (as1 v4i*, as3 v4i*, imm offset, imm cpol).
#if defined(__gfx1250__) && __has_builtin(__builtin_amdgcn_global_load_async_to_lds_b128) && __has_builtin(__builtin_amdgcn_s_wait_asynccnt)
#define HAVE_ASYNC 1
// generic->as(1)/as(3) via integer round-trip (generic LDS addr low 32 bits = LDS offset)
#define GCAST(p) ((__attribute__((address_space(1))) v4i*)(unsigned long long)(p))
#define LCAST(p) ((__attribute__((address_space(3))) v4i*)(unsigned)(unsigned long long)(p))
#else
#define HAVE_ASYNC 0
#endif

// ---------------------------------------------------------------- f32 -> f16
__global__ __launch_bounds__(256) void cvt_f32_f16(const float* __restrict__ src,
                                                   _Float16* __restrict__ dst, long n) {
  long i = (long)blockIdx.x * blockDim.x + threadIdx.x;
  long stride = (long)gridDim.x * blockDim.x;
  for (; i < n; i += stride) dst[i] = (_Float16)src[i];
}

// ------------------------------------------------- WMMA GEMM: C = A(f16) * B(f16)
// A: MxK row-major, B: KxN row-major, C: MxN f32. Double-buffered LDS;
// A tile staged via async-to-LDS when available, B tile VGPR-pipelined + transposed.
__global__ __launch_bounds__(256) void gemm_f16f32(const _Float16* __restrict__ A,
                                                   const _Float16* __restrict__ Bm,
                                                   float* __restrict__ C,
                                                   int M, int N, int K) {
  __shared__ _Float16 lA [2][128][40];   // [m][k]  row stride 80B (16B multiple)
  __shared__ _Float16 lBt[2][128][40];   // [n][k]  transposed B tile
  const int tid  = threadIdx.x;
  const int lane = tid & 31, wave = tid >> 5;
  const int wm = wave & 3, wn = wave >> 2;        // 4 waves in M, 2 in N
  const int hl = lane >> 4, ln = lane & 15;
  const int rowg0 = blockIdx.y * 128;
  const int colg0 = blockIdx.x * 128;

  CFrag acc[2][4];
  for (int a = 0; a < 2; a++)
    for (int b = 0; b < 4; b++)
      for (int r = 0; r < 8; r++) acc[a][b].e[r] = 0.f;

#if !HAVE_ASYNC
  uint4 aReg[2];
#endif
  Pack8 bReg[2];

  auto issueA = [&](int buf, int kk0) {
#pragma unroll
    for (int i = 0; i < 2; i++) {
      int c = tid + i * 256;
      int r = c >> 2, k8 = (c & 3) << 3;
#if HAVE_ASYNC
      __builtin_amdgcn_global_load_async_to_lds_b128(
          GCAST(A + (size_t)(rowg0 + r) * K + kk0 + k8),
          LCAST(&lA[buf][r][k8]), 0, 0);
#else
      aReg[i] = *(const uint4*)(A + (size_t)(rowg0 + r) * K + kk0 + k8);
#endif
    }
  };
  auto loadB = [&](int kk0) {
#pragma unroll
    for (int i = 0; i < 2; i++) {
      int c = tid + i * 256;
      int kr = c >> 4, n8 = (c & 15) << 3;
      bReg[i].u = *(const uint4*)(Bm + (size_t)(kk0 + kr) * N + colg0 + n8);
    }
  };
  auto storeTile = [&](int buf) {
#pragma unroll
    for (int i = 0; i < 2; i++) {
      int c = tid + i * 256;
#if !HAVE_ASYNC
      { int r = c >> 2, k8 = (c & 3) << 3;
        *(uint4*)&lA[buf][r][k8] = aReg[i]; }
#endif
      int kr = c >> 4, n8 = (c & 15) << 3;
#pragma unroll
      for (int e = 0; e < 8; e++) lBt[buf][n8 + e][kr] = bReg[i].h[e];
    }
  };

  issueA(0, 0);
  loadB(0);
  int buf = 0;
  for (int kk0 = 0; kk0 < K; kk0 += 32) {
    storeTile(buf);
#if HAVE_ASYNC
    __builtin_amdgcn_s_wait_asynccnt(0);     // A tile for `buf` resident in LDS
#endif
    __syncthreads();                         // publish buf
    if (kk0 + 32 < K) {                      // issue next tile; latency hides under WMMAs
      issueA(buf ^ 1, kk0 + 32);
      loadB(kk0 + 32);
    }

    AFrag af[2];
#pragma unroll
    for (int mi = 0; mi < 2; mi++) {
      int row = wm * 32 + mi * 16 + ln;
#pragma unroll
      for (int j = 0; j < 8; j++) {
        int kb = ((j < 4) ? 2 * j : 16 + 2 * (j - 4)) + 8 * hl;  // ISA A-layout
        af[mi].e[2 * j]     = lA[buf][row][kb];
        af[mi].e[2 * j + 1] = lA[buf][row][kb + 1];
      }
    }
#pragma unroll
    for (int ni = 0; ni < 4; ni++) {
      AFrag bf;
      int col = wn * 64 + ni * 16 + ln;
#pragma unroll
      for (int t = 0; t < 8; t++) {
        int kb = ((t < 4) ? 2 * t : 16 + 2 * (t - 4)) + 8 * hl;  // ISA B-layout
        bf.e[2 * t]     = lBt[buf][col][kb];
        bf.e[2 * t + 1] = lBt[buf][col][kb + 1];
      }
#pragma unroll
      for (int mi = 0; mi < 2; mi++)
        acc[mi][ni].v = __builtin_amdgcn_wmma_f32_16x16x32_f16(
            false, af[mi].v, false, bf.v, (short)0, acc[mi][ni].v, false, false);
    }
    buf ^= 1;
  }

#pragma unroll
  for (int mi = 0; mi < 2; mi++)
#pragma unroll
    for (int ni = 0; ni < 4; ni++)
#pragma unroll
      for (int r = 0; r < 8; r++) {
        int rg = rowg0 + wm * 32 + mi * 16 + r + 8 * hl;
        int cg = colg0 + wn * 64 + ni * 16 + ln;
        C[(size_t)rg * N + cg] = acc[mi][ni].e[r];
      }
}

// ---------------------- RoPE + relayout [B,S,H,D] f32 -> [B,H,S,D] f16 (scale folded)
__global__ __launch_bounds__(256) void rope_reshape(const float* __restrict__ src,
                                                    const int* __restrict__ pos,
                                                    _Float16* __restrict__ dst,
                                                    int H, float scale) {
  long n = (long)B_ * H * S_ * 64;
  long idx = (long)blockIdx.x * 256 + threadIdx.x;
  long stride = (long)gridDim.x * 256;
  for (; idx < n; idx += stride) {
    int i = (int)(idx & 63);
    long t = idx >> 6;
    int s = (int)(t % S_); t /= S_;
    int h = (int)(t % H);  int b = (int)(t / H);
    float p = (float)pos[b * S_ + s];
    float ang = p * __expf(-0.14391156463f * (float)i);  // 10000^(-i/64)
    float sn, cs; __sincosf(ang, &sn, &cs);
    const float* row = src + ((size_t)b * S_ + s) * ((size_t)H * D_) + (size_t)h * D_;
    float q1 = row[i], q2 = row[i + 64];
    _Float16* orow = dst + (((size_t)b * H + h) * S_ + s) * D_;
    orow[i]      = (_Float16)((q1 * cs - q2 * sn) * scale);
    orow[i + 64] = (_Float16)((q2 * cs + q1 * sn) * scale);
  }
}

// ---------------------- V relayout [B,S,Hkv,D] f32 -> [B,Hkv,S,D] f16
__global__ __launch_bounds__(256) void v_reshape(const float* __restrict__ src,
                                                 _Float16* __restrict__ dst) {
  long n = (long)B_ * HKV_ * S_ * D_;
  long idx = (long)blockIdx.x * 256 + threadIdx.x;
  long stride = (long)gridDim.x * 256;
  for (; idx < n; idx += stride) {
    int d = (int)(idx & (D_ - 1));
    long t = idx >> 7;
    int s = (int)(t % S_); t /= S_;
    int h = (int)(t % HKV_); int b = (int)(t / HKV_);
    dst[(((size_t)b * HKV_ + h) * S_ + s) * D_ + d] =
        (_Float16)src[((size_t)b * S_ + s) * (HKV_ * D_) + h * D_ + d];
  }
}

// ---------------------- causal flash attention, WMMA, online softmax
// grid: (S/128, HQ, B); block 256 = 8 waves, each wave owns 16 query rows.
// K/V tiles double-buffered; next tile's global loads overlap the WMMA bursts.
__global__ __launch_bounds__(256) void flash_attn(const _Float16* __restrict__ Q,
                                                  const _Float16* __restrict__ Kh,
                                                  const _Float16* __restrict__ Vh,
                                                  _Float16* __restrict__ O) {
  __shared__ _Float16 Kt[2][32][136];   // [key][d]   row stride 272B (16B mult.)
  __shared__ _Float16 Vt[2][128][40];   // [d][key]   transposed V tile
  __shared__ _Float16 Ps[8][16][36];    // per-wave P scratch (C-layout -> A-layout)
  const int tid = threadIdx.x, lane = tid & 31, w = tid >> 5;
  const int hl = lane >> 4, ln = lane & 15;
  const int b = blockIdx.z, h = blockIdx.y, qblk = blockIdx.x * 128;
  const int hk = h >> 2;                // GQA 4:1
  const int q0 = qblk + w * 16;

  // persistent Q A-fragments (1/sqrt(D) already folded in)
  AFrag qf[4];
  {
    const _Float16* qp = Q + (((size_t)b * HQ_ + h) * S_ + (q0 + ln)) * D_;
#pragma unroll
    for (int i = 0; i < 4; i++)
#pragma unroll
      for (int j = 0; j < 8; j++) {
        int d0 = ((j < 4) ? 2 * j : 16 + 2 * (j - 4)) + 8 * hl + 32 * i;
        qf[i].e[2 * j] = qp[d0]; qf[i].e[2 * j + 1] = qp[d0 + 1];
      }
  }
  CFrag o[8];
  float mrow[8], lrow[8];
  for (int ni = 0; ni < 8; ni++)
    for (int r = 0; r < 8; r++) o[ni].e[r] = 0.f;
  for (int r = 0; r < 8; r++) { mrow[r] = -1e30f; lrow[r] = 0.f; }

  const int nkb = (qblk + 128) / 32;    // block-uniform (causal handled by masking)
  const _Float16* kbase = Kh + ((size_t)b * HKV_ + hk) * S_ * D_;
  const _Float16* vbase = Vh + ((size_t)b * HKV_ + hk) * S_ * D_;

  uint4 kReg[2]; Pack8 vReg[2];
  auto loadKV = [&](int k0) {
#pragma unroll
    for (int i = 0; i < 2; i++) {
      int c = tid + i * 256;
      int key = c >> 4, d8 = (c & 15) << 3;
      kReg[i]   = *(const uint4*)(kbase + (size_t)(k0 + key) * D_ + d8);
      vReg[i].u = *(const uint4*)(vbase + (size_t)(k0 + key) * D_ + d8);
    }
  };
  auto storeKV = [&](int buf) {
#pragma unroll
    for (int i = 0; i < 2; i++) {
      int c = tid + i * 256;
      int key = c >> 4, d8 = (c & 15) << 3;
      *(uint4*)&Kt[buf][key][d8] = kReg[i];
#pragma unroll
      for (int e = 0; e < 8; e++) Vt[buf][d8 + e][key] = vReg[i].h[e];
    }
  };

  loadKV(0);
  int cur = 0;
  for (int kb = 0; kb < nkb; kb++) {
    const int k0 = kb * 32;
    storeKV(cur);
    __syncthreads();                       // publish tile `cur`
    if (kb + 1 < nkb) loadKV(k0 + 32);     // overlap next tile with WMMAs below

    // scores: S[16q x 32k] = Q . K^T  (two 16-col C-fragments)
    CFrag c0, c1;
    for (int r = 0; r < 8; r++) { c0.e[r] = 0.f; c1.e[r] = 0.f; }
#pragma unroll
    for (int i = 0; i < 4; i++) {
      AFrag b0, b1;
#pragma unroll
      for (int t = 0; t < 8; t++) {
        int dd = 32 * i + ((t < 4) ? 2 * t : 16 + 2 * (t - 4)) + 8 * hl;
        b0.e[2 * t] = Kt[cur][ln][dd];        b0.e[2 * t + 1] = Kt[cur][ln][dd + 1];
        b1.e[2 * t] = Kt[cur][16 + ln][dd];   b1.e[2 * t + 1] = Kt[cur][16 + ln][dd + 1];
      }
      c0.v = __builtin_amdgcn_wmma_f32_16x16x32_f16(false, qf[i].v, false, b0.v,
                                                    (short)0, c0.v, false, false);
      c1.v = __builtin_amdgcn_wmma_f32_16x16x32_f16(false, qf[i].v, false, b1.v,
                                                    (short)0, c1.v, false, false);
    }

    // causal mask + online softmax (16-lane row reductions per wave32 half)
    float p0[8], p1[8], alpha[8];
#pragma unroll
    for (int r = 0; r < 8; r++) {
      int qg = q0 + r + 8 * hl;
      float s0 = ((k0 + ln)      > qg) ? -1e30f : c0.e[r];
      float s1 = ((k0 + 16 + ln) > qg) ? -1e30f : c1.e[r];
      float mx = fmaxf(s0, s1);
#pragma unroll
      for (int off = 1; off < 16; off <<= 1)
        mx = fmaxf(mx, __shfl_xor(mx, off, 16));
      float mnew = fmaxf(mrow[r], mx);
      alpha[r] = __expf(mrow[r] - mnew);
      p0[r] = __expf(s0 - mnew);
      p1[r] = __expf(s1 - mnew);
      float ps = p0[r] + p1[r];
#pragma unroll
      for (int off = 1; off < 16; off <<= 1)
        ps += __shfl_xor(ps, off, 16);
      lrow[r] = lrow[r] * alpha[r] + ps;
      mrow[r] = mnew;
    }
#pragma unroll
    for (int ni = 0; ni < 8; ni++)
#pragma unroll
      for (int r = 0; r < 8; r++) o[ni].e[r] *= alpha[r];

    // P: C-layout -> f16 A-layout via per-wave LDS scratch
#pragma unroll
    for (int r = 0; r < 8; r++) {
      Ps[w][r + 8 * hl][ln]      = (_Float16)p0[r];
      Ps[w][r + 8 * hl][16 + ln] = (_Float16)p1[r];
    }
    __syncthreads();
    AFrag pf;
#pragma unroll
    for (int j = 0; j < 8; j++) {
      int kbp = ((j < 4) ? 2 * j : 16 + 2 * (j - 4)) + 8 * hl;
      pf.e[2 * j] = Ps[w][ln][kbp]; pf.e[2 * j + 1] = Ps[w][ln][kbp + 1];
    }
    // O += P . V  (8 WMMAs cover D=128)
#pragma unroll
    for (int ni = 0; ni < 8; ni++) {
      AFrag vf;
#pragma unroll
      for (int t = 0; t < 8; t++) {
        int kk = ((t < 4) ? 2 * t : 16 + 2 * (t - 4)) + 8 * hl;
        vf.e[2 * t]     = Vt[cur][ni * 16 + ln][kk];
        vf.e[2 * t + 1] = Vt[cur][ni * 16 + ln][kk + 1];
      }
      o[ni].v = __builtin_amdgcn_wmma_f32_16x16x32_f16(false, pf.v, false, vf.v,
                                                       (short)0, o[ni].v, false, false);
    }
    cur ^= 1;
  }

  // normalize and write attn in [B,S,Hq*D] f16 (feeds final GEMM directly)
#pragma unroll
  for (int ni = 0; ni < 8; ni++)
#pragma unroll
    for (int r = 0; r < 8; r++) {
      int srow = q0 + r + 8 * hl;
      float val = o[ni].e[r] / lrow[r];
      O[((size_t)b * S_ + srow) * (HQ_ * D_) + h * D_ + ni * 16 + ln] = (_Float16)val;
    }
}

// ------------------------------------------------------------------ launcher
extern "C" void kernel_launch(void* const* d_in, const int* in_sizes, int n_in,
                              void* d_out, int out_size, void* d_ws, size_t ws_size,
                              hipStream_t stream) {
  const float* X   = (const float*)d_in[0];
  const int*   pos = (const int*)d_in[1];
  const float* Wq  = (const float*)d_in[2];
  const float* Wk  = (const float*)d_in[3];
  const float* Wv  = (const float*)d_in[4];
  const float* Wo  = (const float*)d_in[5];
  float* out = (float*)d_out;

  char* ws = (char*)d_ws;
  size_t off = 0;
  auto alloc = [&](size_t bytes) -> void* {
    void* p = ws + off;
    off += (bytes + 255) & ~(size_t)255;
    return p;
  };
  const size_t NTOK = (size_t)B_ * S_;            // 4096
  _Float16* Xh  = (_Float16*)alloc(NTOK * HID_ * 2);
  _Float16* Wqh = (_Float16*)alloc((size_t)HID_ * HID_ * 2);
  _Float16* Wkh = (_Float16*)alloc((size_t)HID_ * (HKV_ * D_) * 2);
  _Float16* Wvh = (_Float16*)alloc((size_t)HID_ * (HKV_ * D_) * 2);
  _Float16* Woh = (_Float16*)alloc((size_t)HID_ * HID_ * 2);
  float*    Qf  = (float*)alloc(NTOK * HID_ * 4);
  float*    Kf  = (float*)alloc(NTOK * (HKV_ * D_) * 4);
  float*    Vf  = (float*)alloc(NTOK * (HKV_ * D_) * 4);
  _Float16* Qh  = (_Float16*)alloc(NTOK * HID_ * 2);
  _Float16* Khh = (_Float16*)alloc(NTOK * (HKV_ * D_) * 2);
  _Float16* Vhh = (_Float16*)alloc(NTOK * (HKV_ * D_) * 2);
  _Float16* Ah  = (_Float16*)alloc(NTOK * HID_ * 2);

  // 1) precision conversion
  cvt_f32_f16<<<4096, 256, 0, stream>>>(X,  Xh,  (long)NTOK * HID_);
  cvt_f32_f16<<<4096, 256, 0, stream>>>(Wq, Wqh, (long)HID_ * HID_);
  cvt_f32_f16<<<2048, 256, 0, stream>>>(Wk, Wkh, (long)HID_ * HKV_ * D_);
  cvt_f32_f16<<<2048, 256, 0, stream>>>(Wv, Wvh, (long)HID_ * HKV_ * D_);
  cvt_f32_f16<<<4096, 256, 0, stream>>>(Wo, Woh, (long)HID_ * HID_);

  // 2) QKV projections (WMMA)
  gemm_f16f32<<<dim3(32, 32), 256, 0, stream>>>(Xh, Wqh, Qf, 4096, 4096, 4096);
  gemm_f16f32<<<dim3(8,  32), 256, 0, stream>>>(Xh, Wkh, Kf, 4096, 1024, 4096);
  gemm_f16f32<<<dim3(8,  32), 256, 0, stream>>>(Xh, Wvh, Vf, 4096, 1024, 4096);

  // 3) RoPE + relayout (1/sqrt(128) folded into Q)
  rope_reshape<<<8192, 256, 0, stream>>>(Qf, pos, Qh, HQ_, 0.08838834764831845f);
  rope_reshape<<<4096, 256, 0, stream>>>(Kf, pos, Khh, HKV_, 1.0f);
  v_reshape<<<4096, 256, 0, stream>>>(Vf, Vhh);

  // 4) causal flash attention (WMMA)
  flash_attn<<<dim3(S_ / 128, HQ_, B_), 256, 0, stream>>>(Qh, Khh, Vhh, Ah);

  // 5) output projection (WMMA) straight into d_out (f32)
  gemm_f16f32<<<dim3(32, 32), 256, 0, stream>>>(Ah, Woh, out, 4096, 4096, 4096);
}
